// SUN_54013508715191
// MI455X (gfx1250) — compile-verified, matrix operand order checked by reading, not run on hardware
//
#include <hip/hip_runtime.h>
#include <hip/hip_bf16.h>
#include <math.h>

// ---------------------------------------------------------------------------
// SUN 2-layer GNN for MI455X (gfx1250, wave32, WMMA).
//   local = segment_sum(x[dst] -> src)            (atomic scatter, f32)
//   h     = relu(x@W1r + local@W1n + mean(x)@W1g + b1r+b1n+b1g)
//   out   = log_softmax(h@W2r + local2@W2n + mean(h)@W2g + b2*)
// GEMMs use v_wmma_f32_16x16x32_f16 (f16 A/B, f32 accumulate).
// ---------------------------------------------------------------------------

typedef _Float16 h8  __attribute__((ext_vector_type(8)));
typedef _Float16 h16 __attribute__((ext_vector_type(16)));
typedef float    f8  __attribute__((ext_vector_type(8)));

#define FDIM 128
#define CLSN 40
#define PADK 136   // LDS row stride in halves (272B): conflict-free b128 frag loads

union HFrag { h16 v; h8 p[2]; };

// ------------------------------ utility kernels ----------------------------

__global__ void k_zero(float* __restrict__ p, size_t n) {
  size_t i  = (size_t)blockIdx.x * blockDim.x + threadIdx.x;
  size_t st = (size_t)gridDim.x * blockDim.x;
  for (; i < n; i += st) p[i] = 0.0f;
}

// column sums of a [N,128] matrix (for the mean-feature global term)
__global__ void k_colsum(const float* __restrict__ x, float* __restrict__ cs, int N) {
  const int c = threadIdx.x;           // 128 threads
  float s = 0.0f;
  for (int r = blockIdx.x; r < N; r += gridDim.x)
    s += x[(size_t)r * FDIM + c];
  atomicAdd(&cs[c], s);
}

// c[j] = (colsum/N) @ Wg[:,j] + bg[j] + br[j] + bn[j]   (j < M, pad zeros to Mpad)
__global__ void k_cvec(const float* __restrict__ cs, const float* __restrict__ Wg,
                       const float* __restrict__ bg, const float* __restrict__ br,
                       const float* __restrict__ bn, float* __restrict__ cout,
                       int K, int M, int Mpad, float invN) {
  const int j = threadIdx.x;
  if (j >= Mpad) return;
  if (j < M) {
    float s = 0.0f;
    for (int k = 0; k < K; ++k) s += cs[k] * Wg[(size_t)k * M + j];
    cout[j] = s * invN + bg[j] + br[j] + bn[j];
  } else {
    cout[j] = 0.0f;
  }
}

// acc[src[e]][:] += feat[dst[e]][:]   — one wave per edge, float4 per lane
__global__ void k_scatter(const float* __restrict__ feat, const int* __restrict__ src,
                          const int* __restrict__ dst, float* __restrict__ acc, int E) {
  const int e = blockIdx.x * (blockDim.x >> 5) + (threadIdx.x >> 5);
  if (e >= E) return;
  const int lane = threadIdx.x & 31;
  const int s = src[e], d = dst[e];
  const float4 v = ((const float4*)(feat + (size_t)d * FDIM))[lane];
  float* a = acc + (size_t)s * FDIM + lane * 4;
  atomicAdd(a + 0, v.x);
  atomicAdd(a + 1, v.y);
  atomicAdd(a + 2, v.z);
  atomicAdd(a + 3, v.w);
}

// ------------------------------ layer 1 GEMM -------------------------------
// h = relu(x@W1r + local@W1n + c1); also accumulates colsum(h) for layer 2.
// 256 threads = 8 waves; wave w owns output N-tile w (16 cols); grid-stride M.
__global__ __launch_bounds__(256) void k_gemm1(
    const float* __restrict__ x, const float* __restrict__ loc,
    const float* __restrict__ Wr, const float* __restrict__ Wn,
    const float* __restrict__ c1, float* __restrict__ h,
    float* __restrict__ colsumH, int N) {
  extern __shared__ char smem[];
  _Float16* sWr = (_Float16*)smem;            // [128][PADK] transposed: sWr[n][k]
  _Float16* sWn = sWr + FDIM * PADK;
  _Float16* sA1 = sWn + FDIM * PADK;          // [16][PADK]  x tile
  _Float16* sA2 = sA1 + 16 * PADK;            // [16][PADK]  local tile
  float*    sC  = (float*)(sA2 + 16 * PADK);  // [128] fused bias

  const int tid = threadIdx.x;
  for (int i = tid; i < FDIM * FDIM; i += 256) {
    const int k = i >> 7, n = i & 127;
    sWr[n * PADK + k] = (_Float16)Wr[i];
    sWn[n * PADK + k] = (_Float16)Wn[i];
  }
  if (tid < FDIM) sC[tid] = c1[tid];
  __syncthreads();

  const int lane = tid & 31;
  const int wv   = tid >> 5;                  // n-tile index 0..7
  const int mr   = lane & 15;
  const int ko   = (lane < 16) ? 0 : 8;       // lane-half k offset (ISA layout)
  const int nrow = wv * 16 + mr;              // global output column this lane owns
  const int mb   = (lane < 16) ? 0 : 8;       // C/D: M = r + 8*(lane>=16)

  const int ntiles = N >> 4;
  for (int t = blockIdx.x; t < ntiles; t += gridDim.x) {
    const size_t rowbase = (size_t)t * 16;
    for (int i = tid; i < 16 * FDIM; i += 256) {
      const int r = i >> 7, c = i & 127;
      sA1[r * PADK + c] = (_Float16)x  [(rowbase + r) * FDIM + c];
      sA2[r * PADK + c] = (_Float16)loc[(rowbase + r) * FDIM + c];
    }
    __syncthreads();

    f8 acc = {0.f, 0.f, 0.f, 0.f, 0.f, 0.f, 0.f, 0.f};
#pragma unroll
    for (int kt = 0; kt < 4; ++kt) {
      const int kb = kt * 32;
      HFrag a1, a2, b1, b2;
      a1.p[0] = *(const h8*)(sA1 + mr * PADK + kb + ko);
      a1.p[1] = *(const h8*)(sA1 + mr * PADK + kb + 16 + ko);
      a2.p[0] = *(const h8*)(sA2 + mr * PADK + kb + ko);
      a2.p[1] = *(const h8*)(sA2 + mr * PADK + kb + 16 + ko);
      b1.p[0] = *(const h8*)(sWr + nrow * PADK + kb + ko);
      b1.p[1] = *(const h8*)(sWr + nrow * PADK + kb + 16 + ko);
      b2.p[0] = *(const h8*)(sWn + nrow * PADK + kb + ko);
      b2.p[1] = *(const h8*)(sWn + nrow * PADK + kb + 16 + ko);
      acc = __builtin_amdgcn_wmma_f32_16x16x32_f16(false, a1.v, false, b1.v,
                                                   (short)0, acc, false, false);
      acc = __builtin_amdgcn_wmma_f32_16x16x32_f16(false, a2.v, false, b2.v,
                                                   (short)0, acc, false, false);
    }

    const float bias = sC[nrow];
    float csum = 0.0f;
#pragma unroll
    for (int r = 0; r < 8; ++r) {
      float v = acc[r] + bias;
      v = v > 0.0f ? v : 0.0f;
      h[(rowbase + mb + r) * FDIM + nrow] = v;
      csum += v;
    }
    atomicAdd(&colsumH[nrow], csum);
    __syncthreads();
  }
}

// ------------------------------ layer 2 GEMM + log_softmax -----------------
// out = log_softmax(h@W2r + local2@W2n + c2); 96 threads = 3 waves (48 cols,
// cols 40..47 are zero padding).
__global__ __launch_bounds__(96) void k_gemm2(
    const float* __restrict__ hin, const float* __restrict__ loc,
    const float* __restrict__ Wr, const float* __restrict__ Wn,
    const float* __restrict__ c2, float* __restrict__ out, int N) {
  extern __shared__ char smem[];
  _Float16* sWr = (_Float16*)smem;            // [48][PADK] transposed
  _Float16* sWn = sWr + 48 * PADK;
  _Float16* sA1 = sWn + 48 * PADK;            // [16][PADK] h tile
  _Float16* sA2 = sA1 + 16 * PADK;            // [16][PADK] local tile
  float*    sC  = (float*)(sA2 + 16 * PADK);  // [48]
  float*    sL  = sC + 48;                    // [16][40] logits

  const int tid = threadIdx.x;
  for (int i = tid; i < 48 * PADK; i += 96) {
    sWr[i] = (_Float16)0.0f;
    sWn[i] = (_Float16)0.0f;
  }
  __syncthreads();
  for (int i = tid; i < FDIM * CLSN; i += 96) {
    const int k = i / CLSN, n = i % CLSN;
    sWr[n * PADK + k] = (_Float16)Wr[i];
    sWn[n * PADK + k] = (_Float16)Wn[i];
  }
  if (tid < 48) sC[tid] = c2[tid];
  __syncthreads();

  const int lane = tid & 31;
  const int wv   = tid >> 5;                  // n-tile 0..2
  const int mr   = lane & 15;
  const int ko   = (lane < 16) ? 0 : 8;
  const int nrow = wv * 16 + mr;              // 0..47
  const int mb   = (lane < 16) ? 0 : 8;

  const int ntiles = N >> 4;
  for (int t = blockIdx.x; t < ntiles; t += gridDim.x) {
    const size_t rowbase = (size_t)t * 16;
    for (int i = tid; i < 16 * FDIM; i += 96) {
      const int r = i >> 7, c = i & 127;
      sA1[r * PADK + c] = (_Float16)hin[(rowbase + r) * FDIM + c];
      sA2[r * PADK + c] = (_Float16)loc[(rowbase + r) * FDIM + c];
    }
    __syncthreads();

    f8 acc = {0.f, 0.f, 0.f, 0.f, 0.f, 0.f, 0.f, 0.f};
#pragma unroll
    for (int kt = 0; kt < 4; ++kt) {
      const int kb = kt * 32;
      HFrag a1, a2, b1, b2;
      a1.p[0] = *(const h8*)(sA1 + mr * PADK + kb + ko);
      a1.p[1] = *(const h8*)(sA1 + mr * PADK + kb + 16 + ko);
      a2.p[0] = *(const h8*)(sA2 + mr * PADK + kb + ko);
      a2.p[1] = *(const h8*)(sA2 + mr * PADK + kb + 16 + ko);
      b1.p[0] = *(const h8*)(sWr + nrow * PADK + kb + ko);
      b1.p[1] = *(const h8*)(sWr + nrow * PADK + kb + 16 + ko);
      b2.p[0] = *(const h8*)(sWn + nrow * PADK + kb + ko);
      b2.p[1] = *(const h8*)(sWn + nrow * PADK + kb + 16 + ko);
      acc = __builtin_amdgcn_wmma_f32_16x16x32_f16(false, a1.v, false, b1.v,
                                                   (short)0, acc, false, false);
      acc = __builtin_amdgcn_wmma_f32_16x16x32_f16(false, a2.v, false, b2.v,
                                                   (short)0, acc, false, false);
    }

    if (nrow < CLSN) {
      const float bias = sC[nrow];
#pragma unroll
      for (int r = 0; r < 8; ++r) sL[(mb + r) * CLSN + nrow] = acc[r] + bias;
    }
    __syncthreads();

    if (tid < 16) {
      float mx = -3.4e38f;
      for (int j = 0; j < CLSN; ++j) {
        const float v = sL[tid * CLSN + j];
        mx = v > mx ? v : mx;
      }
      float s = 0.0f;
      for (int j = 0; j < CLSN; ++j) s += expf(sL[tid * CLSN + j] - mx);
      const float lse = mx + logf(s);
      for (int j = 0; j < CLSN; ++j)
        out[(rowbase + tid) * CLSN + j] = sL[tid * CLSN + j] - lse;
    }
    __syncthreads();
  }
}

// ------------------------------ launch -------------------------------------

extern "C" void kernel_launch(void* const* d_in, const int* in_sizes, int n_in,
                              void* d_out, int out_size, void* d_ws, size_t ws_size,
                              hipStream_t stream) {
  const float* x   = (const float*)d_in[0];
  const int*   ei  = (const int*)d_in[1];
  const float* W1r = (const float*)d_in[2];
  const float* b1r = (const float*)d_in[3];
  const float* W1n = (const float*)d_in[4];
  const float* b1n = (const float*)d_in[5];
  const float* W1g = (const float*)d_in[6];
  const float* b1g = (const float*)d_in[7];
  const float* W2r = (const float*)d_in[8];
  const float* b2r = (const float*)d_in[9];
  const float* W2n = (const float*)d_in[10];
  const float* b2n = (const float*)d_in[11];
  const float* W2g = (const float*)d_in[12];
  const float* b2g = (const float*)d_in[13];
  float* out = (float*)d_out;

  const int N = in_sizes[0] / FDIM;
  const int E = in_sizes[1] / 2;
  const int* src = ei;
  const int* dst = ei + E;

  const size_t NN = (size_t)N * FDIM;
  float* localb  = (float*)d_ws;       // [N,128] scratch (reused for both layers)
  float* hbuf    = localb + NN;        // [N,128] hidden activations
  float* colsumX = hbuf + NN;          // [128]
  float* colsumH = colsumX + FDIM;     // [128]
  float* c1      = colsumH + FDIM;     // [128]
  float* c2      = c1 + FDIM;          // [64] (40 used)

  const int smem1 = (FDIM * PADK * 2 + 16 * PADK * 2) * (int)sizeof(_Float16)
                    + FDIM * (int)sizeof(float);                       // 78848 B
  const int smem2 = (48 * PADK * 2 + 16 * PADK * 2) * (int)sizeof(_Float16)
                    + (48 + 16 * CLSN) * (int)sizeof(float);           // 37568 B

  const float invN = 1.0f / (float)N;
  const int eblocks = (E + 7) / 8;

  // --- layer 1 ---
  k_zero<<<2048, 256, 0, stream>>>(localb, NN);
  k_zero<<<1, 256, 0, stream>>>(colsumX, 256);        // colsumX + colsumH
  k_colsum<<<512, 128, 0, stream>>>(x, colsumX, N);
  k_cvec<<<1, 128, 0, stream>>>(colsumX, W1g, b1g, b1r, b1n, c1, FDIM, FDIM, FDIM, invN);
  k_scatter<<<eblocks, 256, 0, stream>>>(x, src, dst, localb, E);
  k_gemm1<<<640, 256, smem1, stream>>>(x, localb, W1r, W1n, c1, hbuf, colsumH, N);

  // --- layer 2 ---
  k_zero<<<2048, 256, 0, stream>>>(localb, NN);
  k_cvec<<<1, 128, 0, stream>>>(colsumH, W2g, b2g, b2r, b2n, c2, FDIM, CLSN, 64, invN);
  k_scatter<<<eblocks, 256, 0, stream>>>(hbuf, src, dst, localb, E);
  k_gemm2<<<640, 96, smem2, stream>>>(hbuf, localb, W2r, W2n, c2, out, N);
}
